// Model_83270825935015
// MI455X (gfx1250) — compile-verified, matrix operand order checked by reading, not run on hardware
//
#include <hip/hip_runtime.h>
#include <math.h>

#define Bx   8
#define Lx   720
#define Nx   321
#define KPAD 352              // Nx padded up to multiple of 32 for WMMA K
#define DM   512
#define Hh   8
#define DH   64
#define EL   2
#define PR   336
#define DF   128
#define Ux   35
#define TOK  (Bx*Lx)          // 5760 = 45 * 128
#define EPSF 1e-05f

typedef __attribute__((ext_vector_type(16))) __bf16 v16bf;
typedef __attribute__((ext_vector_type(2)))  __bf16 v2bf;
typedef __attribute__((ext_vector_type(8)))  float  v8f;

// ---------------------------------------------------------------------------
// bf16-WMMA GEMM: C[M,N] = act(A[M,K] @ W[K,N] + bias)
// 256 threads = 8 waves (4x2); block tile 128x128; wave tile 32x64
// => 8 x v_wmma_f32_16x16x32_bf16 per K-step per wave
// Fast path (uniform branch): K%32==0, full M/N tiles -> float4 loads,
// packed cvt, wide LDS stores. Slow path only for the N=336 tail tile.
// ---------------------------------------------------------------------------
__global__ void __launch_bounds__(256)
k_gemm(const float* __restrict__ A, const float* __restrict__ W,
       const float* __restrict__ bias, float* __restrict__ C,
       int M, int N, int K, int act)
{
    __shared__ __align__(32) __bf16 sA[128][32];   // [m][k]
    __shared__ __align__(32) __bf16 sB[128][32];   // [n][k] (transposed stage)

    const int t    = threadIdx.x;
    const int bm   = blockIdx.x;
    const int bn   = blockIdx.y;
    const int wave = t >> 5;
    const int lane = t & 31;
    const int wm   = wave & 3;         // 4 wave-rows  (128/32)
    const int wn   = wave >> 2;        // 2 wave-cols  (128/64)
    const bool fast = ((bn * 128 + 128) <= N) && ((bm * 128 + 128) <= M) && ((K & 31) == 0);

    v8f acc[2][4];
#pragma unroll
    for (int i = 0; i < 2; ++i)
#pragma unroll
        for (int j = 0; j < 4; ++j)
            acc[i][j] = (v8f){0.f,0.f,0.f,0.f,0.f,0.f,0.f,0.f};

    const int ksteps = (K + 31) >> 5;
    for (int kt = 0; kt < ksteps; ++kt) {
        const int k0 = kt << 5;
        if (fast) {
            // ---- A tile 128x32: thread -> (row = t>>1, k-half = (t&1)*16) ----
            const int ar = t >> 1, ak = (t & 1) << 4;
            const float* ap = A + (size_t)(bm * 128 + ar) * K + k0 + ak;
            float4 f0 = *(const float4*)(ap);
            float4 f1 = *(const float4*)(ap + 4);
            float4 f2 = *(const float4*)(ap + 8);
            float4 f3 = *(const float4*)(ap + 12);
            v16bf hv;
            hv[0]  = (__bf16)f0.x; hv[1]  = (__bf16)f0.y; hv[2]  = (__bf16)f0.z; hv[3]  = (__bf16)f0.w;
            hv[4]  = (__bf16)f1.x; hv[5]  = (__bf16)f1.y; hv[6]  = (__bf16)f1.z; hv[7]  = (__bf16)f1.w;
            hv[8]  = (__bf16)f2.x; hv[9]  = (__bf16)f2.y; hv[10] = (__bf16)f2.z; hv[11] = (__bf16)f2.w;
            hv[12] = (__bf16)f3.x; hv[13] = (__bf16)f3.y; hv[14] = (__bf16)f3.z; hv[15] = (__bf16)f3.w;
            *(v16bf*)&sA[ar][ak] = hv;
            // ---- B tile 32x128 -> sB[n][k]: thread covers 2 (k-pair, n-quad) blocks ----
#pragma unroll
            for (int j = 0; j < 2; ++j) {
                int id = t + 256 * j;             // 0..511
                int kp = id >> 5;                 // k pair 0..15
                int nq = (id & 31) << 2;          // n quad 0..124
                const float* wp = W + (size_t)(k0 + 2 * kp) * N + bn * 128 + nq;
                float4 r0 = *(const float4*)(wp);
                float4 r1 = *(const float4*)(wp + N);
                v2bf p0 = { (__bf16)r0.x, (__bf16)r1.x };
                v2bf p1 = { (__bf16)r0.y, (__bf16)r1.y };
                v2bf p2 = { (__bf16)r0.z, (__bf16)r1.z };
                v2bf p3 = { (__bf16)r0.w, (__bf16)r1.w };
                *(v2bf*)&sB[nq + 0][2 * kp] = p0;
                *(v2bf*)&sB[nq + 1][2 * kp] = p1;
                *(v2bf*)&sB[nq + 2][2 * kp] = p2;
                *(v2bf*)&sB[nq + 3][2 * kp] = p3;
            }
            if (kt + 1 < ksteps) {   // L2 prefetch of next K tile
                __builtin_prefetch(ap + 32, 0, 3);
                __builtin_prefetch(W + (size_t)(k0 + 32 + (t >> 3)) * N + bn * 128 + ((t & 7) << 4), 0, 3);
            }
        } else {
            // ---- guarded scalar staging (N tail tile only) ----
#pragma unroll
            for (int j = 0; j < 16; ++j) {
                int idx = t + 256 * j;
                int r = idx >> 5, kk = idx & 31;
                int gm = bm * 128 + r, gk = k0 + kk;
                float val = (gm < M && gk < K) ? A[(size_t)gm * K + gk] : 0.f;
                sA[r][kk] = (__bf16)val;
            }
#pragma unroll
            for (int j = 0; j < 16; ++j) {
                int idx = t + 256 * j;
                int kk = idx >> 7, nn = idx & 127;
                int gk = k0 + kk, gn = bn * 128 + nn;
                float val = (gk < K && gn < N) ? W[(size_t)gk * N + gn] : 0.f;
                sB[nn][kk] = (__bf16)val;
            }
        }
        __syncthreads();

        // ---- fragments: 16 contiguous bf16 along K per lane (k = 16*(lane>>4) + s, same order for A and B) ----
        const int kofs = (lane >> 4) << 4;
        v16bf a0 = *(const v16bf*)&sA[wm * 32 +      (lane & 15)][kofs];
        v16bf a1 = *(const v16bf*)&sA[wm * 32 + 16 + (lane & 15)][kofs];
        v16bf bf[4];
#pragma unroll
        for (int j = 0; j < 4; ++j)
            bf[j] = *(const v16bf*)&sB[wn * 64 + j * 16 + (lane & 15)][kofs];
#pragma unroll
        for (int j = 0; j < 4; ++j) {
            acc[0][j] = __builtin_amdgcn_wmma_f32_16x16x32_bf16(false, a0, false, bf[j], (short)0, acc[0][j], false, false);
            acc[1][j] = __builtin_amdgcn_wmma_f32_16x16x32_bf16(false, a1, false, bf[j], (short)0, acc[1][j], false, false);
        }
        __syncthreads();
    }

    // ---- store: C/D layout = VGPR r -> row r + 8*(lane>=16), col = lane&15 ----
    const int rofs = (lane >> 4) << 3;
    const int cl   = lane & 15;
#pragma unroll
    for (int i = 0; i < 2; ++i)
#pragma unroll
        for (int j = 0; j < 4; ++j)
#pragma unroll
            for (int r = 0; r < 8; ++r) {
                int gm = bm * 128 + wm * 32 + i * 16 + rofs + r;
                int gn = bn * 128 + wn * 64 + j * 16 + cl;
                if (gm < M && gn < N) {
                    float vv = acc[i][j][r];
                    if (bias) vv += bias[gn];
                    if (act == 1) vv = fmaxf(vv, 0.f);
                    C[(size_t)gm * N + gn] = vv;
                }
            }
}

// ---------------------------------------------------------------------------
// Input normalization: per (b,n) mean/std over L (ddof=1); one wave per (b,n)
// Writes xs with padded row stride KPAD.
// ---------------------------------------------------------------------------
__global__ void k_normalize(const float* __restrict__ X, float* __restrict__ XS,
                            float* __restrict__ Mean, float* __restrict__ Std)
{
    int idx  = blockIdx.x * 8 + (threadIdx.x >> 5);
    int lane = threadIdx.x & 31;
    if (idx >= Bx * Nx) return;
    int b = idx / Nx, n = idx % Nx;
    float s = 0.f, s2 = 0.f;
    for (int l = lane; l < Lx; l += 32) {
        float x = X[((size_t)b * Lx + l) * Nx + n];
        s += x; s2 += x * x;
    }
#pragma unroll
    for (int m = 16; m > 0; m >>= 1) { s += __shfl_xor(s, m, 32); s2 += __shfl_xor(s2, m, 32); }
    float mean = s / (float)Lx;
    float var  = fmaxf((s2 - (float)Lx * mean * mean) / (float)(Lx - 1), 0.f);
    float sd   = sqrtf(var) + EPSF;
    float inv  = 1.f / sd;
    for (int l = lane; l < Lx; l += 32) {
        XS[((size_t)b * Lx + l) * KPAD + n] =
            (X[((size_t)b * Lx + l) * Nx + n] - mean) * inv;
    }
    if (lane == 0) { Mean[b * Nx + n] = mean; Std[b * Nx + n] = sd; }
}

// zero the K-pad columns of xs (cols Nx..KPAD-1)
__global__ void k_padxs(float* __restrict__ xs)
{
    int i = blockIdx.x * 256 + threadIdx.x;
    const int PADW = KPAD - Nx;                 // 31
    if (i >= TOK * PADW) return;
    int r = i / PADW, c = Nx + i % PADW;
    xs[(size_t)r * KPAD + c] = 0.f;
}

// padded copy of embed_w: (Nx,DM) -> (KPAD,DM), zero rows Nx..KPAD-1
__global__ void k_padW(const float* __restrict__ w, float* __restrict__ wp)
{
    int i = blockIdx.x * 256 + threadIdx.x;
    if (i >= KPAD * DM) return;
    int r = i >> 9;                             // DM == 512
    wp[i] = (r < Nx) ? w[i] : 0.f;
}

// ---------------------------------------------------------------------------
// ProbSparse M-measure: M[b,h,l] = max_u(q.k_samp) - sum_u(q.k_samp)/L
// ---------------------------------------------------------------------------
__global__ void k_probM(const float* __restrict__ Q, const float* __restrict__ Km,
                        const int* __restrict__ samp, float* __restrict__ Mout)
{
    int idx  = blockIdx.x * 8 + (threadIdx.x >> 5);
    int lane = threadIdx.x & 31;
    if (idx >= Bx * Hh * Lx) return;
    int l = idx % Lx, h = (idx / Lx) % Hh, b = idx / (Lx * Hh);
    const float* qp = &Q[((size_t)b * Lx + l) * DM + h * DH];
    float q0 = qp[lane], q1 = qp[32 + lane];
    float mx = -3.4e38f, sm = 0.f;
    for (int u = 0; u < Ux; ++u) {
        int ks = samp[l * Ux + u];
        const float* kp = &Km[((size_t)b * Lx + ks) * DM + h * DH];
        float p = q0 * kp[lane] + q1 * kp[32 + lane];
#pragma unroll
        for (int m = 16; m > 0; m >>= 1) p += __shfl_xor(p, m, 32);
        mx = fmaxf(mx, p); sm += p;
    }
    if (lane == 0) Mout[(size_t)(b * Hh + h) * Lx + l] = mx - sm / (float)Lx;
}

// ---------------------------------------------------------------------------
// Top-U per (b,h): iterative argmax, first-index tie-break; one wave per (b,h)
// ---------------------------------------------------------------------------
__global__ void k_topu(const float* __restrict__ Mb, int* __restrict__ mtop)
{
    int bh = blockIdx.x, lane = threadIdx.x;
    __shared__ float sM[Lx];
    const float* row = Mb + (size_t)bh * Lx;
    for (int l = lane; l < Lx; l += 32) sM[l] = row[l];
    __syncthreads();
    for (int u = 0; u < Ux; ++u) {
        float bv = -3.4e38f; int bi = 0x7fffffff;
        for (int l = lane; l < Lx; l += 32) {
            float v = sM[l];
            if (v > bv || (v == bv && l < bi)) { bv = v; bi = l; }
        }
#pragma unroll
        for (int m = 16; m > 0; m >>= 1) {
            float ov = __shfl_xor(bv, m, 32);
            int   oi = __shfl_xor(bi, m, 32);
            if (ov > bv || (ov == bv && oi < bi)) { bv = ov; bi = oi; }
        }
        if (lane == 0) { mtop[bh * Ux + u] = bi; sM[bi] = -3.4e38f; }
        __syncthreads();
    }
}

// V column means per (b, c)
__global__ void k_colmean(const float* __restrict__ V, float* __restrict__ cm)
{
    int tid = blockIdx.x * 256 + threadIdx.x;
    if (tid >= Bx * DM) return;
    int b = tid >> 9, c = tid & (DM - 1);
    float s = 0.f;
    for (int l = 0; l < Lx; ++l) s += V[((size_t)b * Lx + l) * DM + c];
    cm[tid] = s / (float)Lx;
}

// ctx[b,l,c] = colmean[b,c]
__global__ void k_fill(float* __restrict__ ctx, const float* __restrict__ cm)
{
    size_t i = (size_t)blockIdx.x * 256 + threadIdx.x;
    if (i >= (size_t)TOK * DM) return;
    int c = (int)(i & (DM - 1));
    int b = (int)(i / ((size_t)Lx * DM));
    ctx[i] = cm[b * DM + c];
}

// ---------------------------------------------------------------------------
// Attention for selected queries: 64 threads per (b,h,u); softmax over L=720
// ---------------------------------------------------------------------------
__global__ void __launch_bounds__(64)
k_attn_sel(const float* __restrict__ Q, const float* __restrict__ Km,
           const float* __restrict__ V, const int* __restrict__ mtop,
           float* __restrict__ ctx)
{
    __shared__ float qs[DH];
    __shared__ float sc[Lx];
    __shared__ float red[64];
    int t = threadIdx.x;
    int u = blockIdx.x % Ux;
    int h = (blockIdx.x / Ux) % Hh;
    int b = blockIdx.x / (Ux * Hh);
    int lsel = mtop[(b * Hh + h) * Ux + u];

    qs[t] = Q[((size_t)b * Lx + lsel) * DM + h * DH + t];
    __syncthreads();

    for (int l = t; l < Lx; l += 64) {
        const float* kp = &Km[((size_t)b * Lx + l) * DM + h * DH];
        float d = 0.f;
#pragma unroll
        for (int e = 0; e < DH; ++e) d += qs[e] * kp[e];
        sc[l] = d * 0.125f;   // 1/sqrt(64)
    }
    __syncthreads();

    float mx = -3.4e38f;
    for (int l = t; l < Lx; l += 64) mx = fmaxf(mx, sc[l]);
    red[t] = mx; __syncthreads();
    for (int s2 = 32; s2 > 0; s2 >>= 1) { if (t < s2) red[t] = fmaxf(red[t], red[t + s2]); __syncthreads(); }
    float mxv = red[0]; __syncthreads();

    float sm = 0.f;
    for (int l = t; l < Lx; l += 64) { float e = expf(sc[l] - mxv); sc[l] = e; sm += e; }
    red[t] = sm; __syncthreads();
    for (int s2 = 32; s2 > 0; s2 >>= 1) { if (t < s2) red[t] += red[t + s2]; __syncthreads(); }
    float inv = 1.f / red[0];
    __syncthreads();

    float acc = 0.f;
    for (int l = 0; l < Lx; ++l) acc += sc[l] * V[((size_t)b * Lx + l) * DM + h * DH + t];
    ctx[((size_t)b * Lx + lsel) * DM + h * DH + t] = acc * inv;
}

// ---------------------------------------------------------------------------
// out = LayerNorm(in1 - in2) * g + b ; one wave per row of 512
// ---------------------------------------------------------------------------
__global__ void k_ln_sub(const float* __restrict__ in1, const float* __restrict__ in2,
                         const float* __restrict__ g, const float* __restrict__ bb,
                         float* __restrict__ out)
{
    int row  = blockIdx.x * 8 + (threadIdx.x >> 5);
    int lane = threadIdx.x & 31;
    if (row >= TOK) return;
    const float* p1 = in1 + (size_t)row * DM;
    const float* p2 = in2 + (size_t)row * DM;
    float d[16]; float s = 0.f;
#pragma unroll
    for (int j = 0; j < 16; ++j) { int c = lane + 32 * j; d[j] = p1[c] - p2[c]; s += d[j]; }
#pragma unroll
    for (int m = 16; m > 0; m >>= 1) s += __shfl_xor(s, m, 32);
    float mu = s / (float)DM;
    float vs = 0.f;
#pragma unroll
    for (int j = 0; j < 16; ++j) { float z = d[j] - mu; vs += z * z; }
#pragma unroll
    for (int m = 16; m > 0; m >>= 1) vs += __shfl_xor(vs, m, 32);
    float rs = rsqrtf(vs / (float)DM + EPSF);
#pragma unroll
    for (int j = 0; j < 16; ++j) {
        int c = lane + 32 * j;
        out[(size_t)row * DM + c] = (d[j] - mu) * rs * g[c] + bb[c];
    }
}

__global__ void k_zero(float* __restrict__ y, int n)
{ int i = blockIdx.x * 256 + threadIdx.x; if (i < n) y[i] = 0.f; }

__global__ void k_gate(float* __restrict__ y, const float* __restrict__ o1,
                       const float* __restrict__ o2, int n)
{
    int i = blockIdx.x * 256 + threadIdx.x;
    if (i < n) y[i] += o1[i] * (1.f / (1.f + expf(-o2[i])));
}

// out[b,p,n] = y[b,n,p] * std[b,n] + mean[b,n]
__global__ void k_final(const float* __restrict__ y, const float* __restrict__ Std,
                        const float* __restrict__ Mean, float* __restrict__ out)
{
    int i = blockIdx.x * 256 + threadIdx.x;
    if (i >= Bx * PR * Nx) return;
    int n = i % Nx, p = (i / Nx) % PR, b = i / (Nx * PR);
    out[i] = y[((size_t)b * Lx + n) * PR + p] * Std[b * Nx + n] + Mean[b * Nx + n];
}

// ---------------------------------------------------------------------------
static inline void gemm(hipStream_t s, const float* A, const float* W, const float* b,
                        float* C, int M, int N, int K, int act)
{
    dim3 g((M + 127) / 128, (N + 127) / 128);
    k_gemm<<<g, 256, 0, s>>>(A, W, b, C, M, N, K, act);
}

extern "C" void kernel_launch(void* const* d_in, const int* in_sizes, int n_in,
                              void* d_out, int out_size, void* d_ws, size_t ws_size,
                              hipStream_t stream)
{
    (void)in_sizes; (void)n_in; (void)out_size; (void)ws_size;
    const float* x_enc  = (const float*)d_in[0];
    const int*   samp   = (const int*)  d_in[1];
    const float* emb_w  = (const float*)d_in[2];
    const float* emb_b  = (const float*)d_in[3];
    const float* wq     = (const float*)d_in[4];
    const float* bq     = (const float*)d_in[5];
    const float* wk     = (const float*)d_in[6];
    const float* bk     = (const float*)d_in[7];
    const float* wv     = (const float*)d_in[8];
    const float* bv     = (const float*)d_in[9];
    const float* wo     = (const float*)d_in[10];
    const float* bo     = (const float*)d_in[11];
    const float* ln1g   = (const float*)d_in[12];
    const float* ln1b   = (const float*)d_in[13];
    const float* w1     = (const float*)d_in[14];
    const float* b1     = (const float*)d_in[15];
    const float* w2     = (const float*)d_in[16];
    const float* b2     = (const float*)d_in[17];
    const float* ln2g   = (const float*)d_in[18];
    const float* ln2b   = (const float*)d_in[19];
    const float* wout   = (const float*)d_in[20];
    const float* bout   = (const float*)d_in[21];
    const float* wg     = (const float*)d_in[22];
    const float* bg     = (const float*)d_in[23];
    float* out = (float*)d_out;

    float* ws = (float*)d_ws;
    size_t off = 0;
    auto alloc = [&](size_t n) { float* p = ws + off; off += (n + 63) & ~(size_t)63; return p; };
    float* xs    = alloc((size_t)TOK * KPAD);       // padded embed input
    float* wpad  = alloc((size_t)KPAD * DM);        // padded embed weights
    float* meanb = alloc((size_t)Bx * Nx);
    float* stdb  = alloc((size_t)Bx * Nx);
    float* h     = alloc((size_t)TOK * DM);
    float* qb    = alloc((size_t)TOK * DM);         // reused as attn_out
    float* kb    = alloc((size_t)TOK * DM);         // reused as hx
    float* vb    = alloc((size_t)TOK * DM);         // reused as ff1
    float* ctx   = alloc((size_t)TOK * DM);         // reused as ff
    float* Mbuf  = alloc((size_t)Bx * Hh * Lx);
    int*   mtop  = (int*)alloc((size_t)Bx * Hh * Ux);
    float* cmean = alloc((size_t)Bx * DM);
    float* o1    = alloc((size_t)TOK * PR);
    float* o2    = alloc((size_t)TOK * PR);
    float* ybuf  = alloc((size_t)TOK * PR);

    // 1) input normalization (+ K padding) and padded embed weights
    k_normalize<<<(Bx * Nx + 7) / 8, 256, 0, stream>>>(x_enc, xs, meanb, stdb);
    k_padxs<<<(TOK * (KPAD - Nx) + 255) / 256, 256, 0, stream>>>(xs);
    k_padW<<<(KPAD * DM + 255) / 256, 256, 0, stream>>>(emb_w, wpad);
    // 2) embedding: h = xs @ wpad + emb_b   (K = 352, fully aligned)
    gemm(stream, xs, wpad, emb_b, h, TOK, DM, KPAD, 0);
    // 3) y = 0
    k_zero<<<(TOK * PR + 255) / 256, 256, 0, stream>>>(ybuf, TOK * PR);

    for (int l = 0; l < EL; ++l) {
        const float* wq_l = wq + (size_t)l * DM * DM; const float* bq_l = bq + (size_t)l * DM;
        const float* wk_l = wk + (size_t)l * DM * DM; const float* bk_l = bk + (size_t)l * DM;
        const float* wv_l = wv + (size_t)l * DM * DM; const float* bv_l = bv + (size_t)l * DM;
        const float* wo_l = wo + (size_t)l * DM * DM; const float* bo_l = bo + (size_t)l * DM;
        const float* w1_l = w1 + (size_t)l * DM * DF; const float* b1_l = b1 + (size_t)l * DF;
        const float* w2_l = w2 + (size_t)l * DF * DM; const float* b2_l = b2 + (size_t)l * DM;
        const float* wo_t = wout + (size_t)l * DM * PR; const float* bo_t = bout + (size_t)l * PR;
        const float* wg_l = wg + (size_t)l * DM * PR;   const float* bg_l = bg + (size_t)l * PR;

        // Q, K, V projections
        gemm(stream, h, wq_l, bq_l, qb, TOK, DM, DM, 0);
        gemm(stream, h, wk_l, bk_l, kb, TOK, DM, DM, 0);
        gemm(stream, h, wv_l, bv_l, vb, TOK, DM, DM, 0);

        // ProbSparse attention
        k_probM<<<(Bx * Hh * Lx) / 8, 256, 0, stream>>>(qb, kb, samp, Mbuf);
        k_topu<<<Bx * Hh, 32, 0, stream>>>(Mbuf, mtop);
        k_colmean<<<(Bx * DM + 255) / 256, 256, 0, stream>>>(vb, cmean);
        k_fill<<<(int)(((size_t)TOK * DM + 255) / 256), 256, 0, stream>>>(ctx, cmean);
        k_attn_sel<<<Bx * Hh * Ux, 64, 0, stream>>>(qb, kb, vb, mtop, ctx);

        // output projection (reuse qb as attn_out)
        gemm(stream, ctx, wo_l, bo_l, qb, TOK, DM, DM, 0);
        // hx = LN(h - attn_out)  (reuse kb)
        k_ln_sub<<<TOK / 8, 256, 0, stream>>>(h, qb, ln1g + (size_t)l * DM, ln1b + (size_t)l * DM, kb);
        // ff = relu(hx @ w1 + b1) @ w2 + b2   (reuse vb for ff1, ctx for ff)
        gemm(stream, kb, w1_l, b1_l, vb, TOK, DF, DM, 1);
        gemm(stream, vb, w2_l, b2_l, ctx, TOK, DM, DF, 0);
        // h = LN(hx - ff)
        k_ln_sub<<<TOK / 8, 256, 0, stream>>>(kb, ctx, ln2g + (size_t)l * DM, ln2b + (size_t)l * DM, h);

        // gated head: y += (h@wout + bout) * sigmoid(h@wg + bg)
        gemm(stream, h, wo_t, bo_t, o1, TOK, PR, DM, 0);
        gemm(stream, h, wg_l, bg_l, o2, TOK, PR, DM, 0);
        k_gate<<<(TOK * PR + 255) / 256, 256, 0, stream>>>(ybuf, o1, o2, TOK * PR);
    }

    // de-normalize + transpose
    k_final<<<(Bx * PR * Nx + 255) / 256, 256, 0, stream>>>(ybuf, stdb, meanb, out);
}
